// LinearLinkPredictor_26843545600130
// MI455X (gfx1250) — matrix operand order repaired and checked by reading, not compiled, not run on hardware
//
#include <hip/hip_runtime.h>
#include <hip/hip_bf16.h>
#include <math.h>

// ---------------------------------------------------------------------------
// LinearLinkPredictor for MI455X (gfx1250, wave32, WMMA).
//
// Restructuring: h = relu(lin_src(src) + lin_dst(dst)) is per-edge, but the
// linears are per-NODE. Precompute 4 node tables with WMMA-bf16 GEMM:
//   A_user = z_user@W_src.T+b_src   B_user = z_user@W_dst.T+b_dst
//   A_item = z_item@W_src.T+b_src   B_item = z_item@W_dst.T+b_dst
// Edge pass (bandwidth bound, ~2 GB of 1KB-row gathers):
//   out[e] = sigmoid( w_fin . relu(A_src[s] + B_dst[d]) + b_fin )
// ---------------------------------------------------------------------------

typedef __attribute__((ext_vector_type(16))) __bf16 v16bf;
typedef __attribute__((ext_vector_type(8)))  float  v8f;

#define DCH 256   // in_channels (= out_channels of the two linears)

// ---------------------------------------------------------------------------
// Phase 1: Out[N,256] = Z[N,256] @ W[256,256]^T + bias  (fp32 in, bf16 WMMA,
// f32 accumulate). One wave computes a 16(M) x 64(N) strip: A frag reused
// across 4 B frags; K=256 fully unrolled in 32-chunks -> 32 v_wmma per wave.
//
// Fragment layouts used here (16-bit, wave32; A/B in contiguous k = 16h+s, not ISA table order):
//  A 16x32: lane L (0..15): M=L, halves 0..15 = K {0..15}
//           lane L (16..31): M=L-16, halves 0..15 = K {16..31}
//  B 32x16: lane L: N=L&15; lanes 0..15 hold K 0..15, lanes 16..31 K 16..31
//  C/D 16x16 f32: VGPR r, lane L: M = (L<16 ? r : r+8), N = L&15
// ---------------------------------------------------------------------------
__global__ __launch_bounds__(256)
void lin_table_kernel(const float* __restrict__ Z,
                      const float* __restrict__ W,
                      const float* __restrict__ bias,
                      float* __restrict__ Out,
                      int nRows)
{
    const int lane  = threadIdx.x & 31;
    const int wave  = threadIdx.x >> 5;
    const int mTiles = (nRows + 15) >> 4;
    const int tile   = blockIdx.x * 8 + wave;      // 8 waves / block
    if (tile >= mTiles * 4) return;                // 4 N-strips of 64 cover D=256

    const int m0 = (tile >> 2) * 16;
    const int n0 = (tile & 3) * 64;

    const int mrow  = m0 + (lane & 15);
    const int mload = (mrow < nRows) ? mrow : (nRows - 1);  // clamp (N%16==0 normally)
    const int aK = (lane >> 4) * 16;   // A-frag per-lane K base inside 32-chunk
    const int bK = (lane >> 4) * 16;   // B-frag per-lane K base inside 32-chunk

    const float* __restrict__ arow = Z + (size_t)mload * DCH;

    v8f acc[4] = {v8f{}, v8f{}, v8f{}, v8f{}};

    #pragma unroll
    for (int k0 = 0; k0 < DCH; k0 += 32) {
        // ---- A fragment: two contiguous 8-float runs -> 2x global_load_b128
        v16bf a;
        {
            const float* ap = arow + k0 + aK;
            #pragma unroll
            for (int h = 0; h < 8; ++h) a[h]     = (__bf16)ap[h];
            #pragma unroll
            for (int h = 0; h < 8; ++h) a[8 + h] = (__bf16)ap[8 + h];
        }
        // ---- 4 B fragments (W rows are L2-resident: W is only 256 KB)
        #pragma unroll
        for (int s = 0; s < 4; ++s) {
            const int ncol = n0 + s * 16 + (lane & 15);
            const float* bp = W + (size_t)ncol * DCH + k0 + bK;  // 16 contiguous floats
            v16bf b;
            #pragma unroll
            for (int h = 0; h < 16; ++h) b[h] = (__bf16)bp[h];
            acc[s] = __builtin_amdgcn_wmma_f32_16x16x32_bf16(
                         /*neg_a=*/false, a, /*neg_b=*/false, b,
                         /*c_mod=*/(short)0, acc[s],
                         /*reuse_a=*/false, /*reuse_b=*/false);
        }
    }

    // ---- epilogue: + bias, store (C/D layout: VGPR r -> M = r or r+8)
    #pragma unroll
    for (int s = 0; s < 4; ++s) {
        const int ncol = n0 + s * 16 + (lane & 15);
        const float bv = bias[ncol];
        #pragma unroll
        for (int r = 0; r < 8; ++r) {
            const int m = m0 + ((lane < 16) ? r : r + 8);
            if (m < nRows) Out[(size_t)m * DCH + ncol] = acc[s][r] + bv;
        }
    }
}

// ---------------------------------------------------------------------------
// Phase 2: one wave per edge. Each lane handles 8 channels (2x float4 loads
// from each 1KB-aligned row), relu+dot, wave32 xor-shuffle reduction, sigmoid.
// Pure gather-bandwidth kernel: ~2 GB total -> ~88 us at 23.3 TB/s.
// ---------------------------------------------------------------------------
__global__ __launch_bounds__(256)
void edge_score_kernel(const float* __restrict__ SrcTab,
                       const float* __restrict__ DstTab,
                       const int*   __restrict__ eidx,   // [2,E] flat
                       const float* __restrict__ wfin,   // [256]
                       const float* __restrict__ bfin,   // [1]
                       float* __restrict__ out,          // [E]
                       int E)
{
    const int lane = threadIdx.x & 31;
    const int e    = blockIdx.x * 8 + (threadIdx.x >> 5);
    if (e >= E) return;

    const int s = eidx[e];
    const int d = eidx[E + e];

    const float4* a4 = (const float4*)(SrcTab + (size_t)s * DCH) + lane * 2;
    const float4* b4 = (const float4*)(DstTab + (size_t)d * DCH) + lane * 2;
    const float4* w4 = (const float4*)wfin + lane * 2;

    float part = 0.f;
    #pragma unroll
    for (int i = 0; i < 2; ++i) {
        const float4 av = a4[i], bv = b4[i], wv = w4[i];
        float h;
        h = av.x + bv.x; part += fmaxf(h, 0.f) * wv.x;
        h = av.y + bv.y; part += fmaxf(h, 0.f) * wv.y;
        h = av.z + bv.z; part += fmaxf(h, 0.f) * wv.z;
        h = av.w + bv.w; part += fmaxf(h, 0.f) * wv.w;
    }
    #pragma unroll
    for (int off = 16; off > 0; off >>= 1)
        part += __shfl_xor(part, off, 32);

    if (lane == 0)
        out[e] = 1.0f / (1.0f + __expf(-(part + bfin[0])));
}

// ---------------------------------------------------------------------------
// Fallback (only if workspace < ~391 MiB): direct per-edge compute,
// one 256-thread block per edge. Slow but self-contained.
// ---------------------------------------------------------------------------
__global__ __launch_bounds__(256)
void edge_direct_kernel(const float* __restrict__ zs_tab,
                        const float* __restrict__ zd_tab,
                        const int*   __restrict__ eidx,
                        const float* __restrict__ Wsrc, const float* __restrict__ bsrc,
                        const float* __restrict__ Wdst, const float* __restrict__ bdst,
                        const float* __restrict__ wfin, const float* __restrict__ bfin,
                        float* __restrict__ out, int E)
{
    __shared__ float ssrc[DCH], sdst[DCH], red[DCH];
    const int e = blockIdx.x;
    if (e >= E) return;
    const int t = threadIdx.x;
    const int s = eidx[e], d = eidx[E + e];
    ssrc[t] = zs_tab[(size_t)s * DCH + t];
    sdst[t] = zd_tab[(size_t)d * DCH + t];
    __syncthreads();

    float h = bsrc[t] + bdst[t];
    const float* w1 = Wsrc + (size_t)t * DCH;
    const float* w2 = Wdst + (size_t)t * DCH;
    #pragma unroll 8
    for (int k = 0; k < DCH; ++k) h += w1[k] * ssrc[k] + w2[k] * sdst[k];
    red[t] = fmaxf(h, 0.f) * wfin[t];
    __syncthreads();
    for (int st = 128; st > 0; st >>= 1) {
        if (t < st) red[t] += red[t + st];
        __syncthreads();
    }
    if (t == 0) out[e] = 1.0f / (1.0f + __expf(-(red[0] + bfin[0])));
}

// ---------------------------------------------------------------------------
// Host launcher
// Inputs: 0 z_user[N,256] f32, 1 z_item[N,256] f32, 2 edge_idx_a[2,E] i32,
//         3 edge_idx_b[2,E] i32, 4 W_src[256,256], 5 b_src[256],
//         6 W_dst[256,256], 7 b_dst[256], 8 W_fin[1,256], 9 b_fin[1]
// Output: [2E] f32 (type-a scores then type-b scores)
// ---------------------------------------------------------------------------
extern "C" void kernel_launch(void* const* d_in, const int* in_sizes, int n_in,
                              void* d_out, int out_size, void* d_ws, size_t ws_size,
                              hipStream_t stream)
{
    const float* z_user = (const float*)d_in[0];
    const float* z_item = (const float*)d_in[1];
    const int*   ea     = (const int*)  d_in[2];
    const int*   eb     = (const int*)  d_in[3];
    const float* W_src  = (const float*)d_in[4];
    const float* b_src  = (const float*)d_in[5];
    const float* W_dst  = (const float*)d_in[6];
    const float* b_dst  = (const float*)d_in[7];
    const float* W_fin  = (const float*)d_in[8];
    const float* b_fin  = (const float*)d_in[9];
    float* out = (float*)d_out;

    const int N = in_sizes[0] / DCH;   // 100000
    const int E = in_sizes[2] / 2;     // 500000

    const size_t tabElems = (size_t)N * DCH;
    const size_t need     = 4 * tabElems * sizeof(float);

    if (ws_size >= need) {
        float* Auser = (float*)d_ws;
        float* Buser = Auser + tabElems;
        float* Aitem = Buser + tabElems;
        float* Bitem = Aitem + tabElems;

        const int mTiles   = (N + 15) >> 4;
        const int gemmBlks = (mTiles * 4 + 7) / 8;   // 8 waves/block, 16x64 per wave
        dim3 blk(256);

        lin_table_kernel<<<gemmBlks, blk, 0, stream>>>(z_user, W_src, b_src, Auser, N);
        lin_table_kernel<<<gemmBlks, blk, 0, stream>>>(z_user, W_dst, b_dst, Buser, N);
        lin_table_kernel<<<gemmBlks, blk, 0, stream>>>(z_item, W_src, b_src, Aitem, N);
        lin_table_kernel<<<gemmBlks, blk, 0, stream>>>(z_item, W_dst, b_dst, Bitem, N);

        const int edgeBlks = (E + 7) / 8;            // 8 edges (waves) per block
        // type a: src = user, dst = item
        edge_score_kernel<<<edgeBlks, blk, 0, stream>>>(Auser, Bitem, ea, W_fin, b_fin, out,     E);
        // type b: src = item, dst = user
        edge_score_kernel<<<edgeBlks, blk, 0, stream>>>(Aitem, Buser, eb, W_fin, b_fin, out + E, E);
    } else {
        edge_direct_kernel<<<E, 256, 0, stream>>>(z_user, z_item, ea,
                                                  W_src, b_src, W_dst, b_dst,
                                                  W_fin, b_fin, out, E);
        edge_direct_kernel<<<E, 256, 0, stream>>>(z_item, z_user, eb,
                                                  W_src, b_src, W_dst, b_dst,
                                                  W_fin, b_fin, out + E, E);
    }
}
